// LogNeuralCDE_28707561406815
// MI455X (gfx1250) — matrix operand order repaired and checked by reading, not compiled, not run on hardware
//
#include <hip/hip_runtime.h>
#include <hip/hip_bf16.h>

// CDNA5 / gfx1250: wave32, WMMA 16x16x32 f16 (fp32 accum).
typedef __attribute__((ext_vector_type(16))) _Float16 v16h;
typedef __attribute__((ext_vector_type(8)))  float    v8f;

#define D_DIM    16
#define H_DIM    128
#define WID      256
#define OUTD     2048      /* D*H */
#define LSIG_N   137
#define NSTEPS   100
#define NTHREADS 512       /* 16 waves = 4 waves/SIMD */

#if __has_builtin(__builtin_amdgcn_sched_barrier)
#define SCHED_FENCE() __builtin_amdgcn_sched_barrier(0)
#else
#define SCHED_FENCE()
#endif
// Phase barrier: workgroup sync + hard fence against cross-phase code motion
// (prevents the scheduler from hoisting next-phase global loads over the
//  barrier, which was causing b128 spill/reload through scratch).
#define PHASE_SYNC() do { SCHED_FENCE(); __syncthreads(); SCHED_FENCE(); } while (0)

union H8    { float4 f;    _Float16 h[8]; };
union HFrag { v16h v;      float4  f[2]; };

// ---------------------------------------------------------------------------
// Prep: fp32 weight [K][N] row-major -> fp16, pre-swizzled into a contiguous-k WMMA
// B-fragment layout: tile (kt,nt), lane = (n&15) + (k in-tile >=16 ? 16 : 0),
// 16 contiguous halves per lane = column n, K = kt*32 + khalf + e.
// ---------------------------------------------------------------------------
__global__ void swz_kernel(const float* __restrict__ w, _Float16* __restrict__ out,
                           int K, int N) {
  int total = K * N;
  int NT = N >> 4;
  for (int id = blockIdx.x * blockDim.x + threadIdx.x; id < total;
       id += gridDim.x * blockDim.x) {
    int k = id / N, n = id - k * N;
    int kt = k >> 5, kr = k & 31, nt = n >> 4, nl = n & 15;
    int lane = nl + (kr & 16);
    int e = kr & 15;
    out[(((size_t)kt * NT + nt) * 32 + lane) * 16 + e] = (_Float16)w[id];
  }
}

// ---------------------------------------------------------------------------
// One 16x16 output tile of  A(16xK) @ B(KxN), K-loop of v_wmma_f32_16x16x32_f16.
// A: LDS row-major [16][K] fp16 (two ds_load_b128 per fragment).
// B: global pre-swizzled (two global_load_b128 per fragment).
// unroll 2 = one fragment set in flight while one feeds the WMMA.
// ---------------------------------------------------------------------------
__device__ __forceinline__ v8f tile_mm(const _Float16* __restrict__ Bswz,
                                       const _Float16* A, int K, int NT,
                                       int nt, int lane) {
  v8f acc = {0.f, 0.f, 0.f, 0.f, 0.f, 0.f, 0.f, 0.f};
  const int m = lane & 15;
  const int abase = (lane < 16) ? 0 : 16;      // K-halves: contiguous, matches B swizzle
  const int nkt = K >> 5;
#pragma unroll 2
  for (int kt = 0; kt < nkt; ++kt) {
    HFrag a, b;
    const _Float16* arow = A + m * K + kt * 32 + abase;
    a.f[0] = *(const float4*)(arow);           // K = base+0..7   (e0..7)
    a.f[1] = *(const float4*)(arow + 8);       // K = base+8..15  (e8..15)
    const _Float16* bp = Bswz + (((size_t)kt * NT + nt) * 32 + lane) * 16;
    b.f[0] = *(const float4*)(bp);
    b.f[1] = *(const float4*)(bp + 8);
    acc = __builtin_amdgcn_wmma_f32_16x16x32_f16(false, a.v, false, b.v,
                                                 (short)0, acc, false, false);
  }
  return acc;
}

// Primal matvec column c using the same swizzled fp16 weights.
__device__ __forceinline__ float dotcol(const _Float16* __restrict__ wswz,
                                        const float* in, int K, int NT, int c) {
  int nt = c >> 4, n = c & 15;
  float acc = 0.f;
  int nkt = K >> 5;
#pragma unroll 2
  for (int kt = 0; kt < nkt; ++kt) {
    const _Float16* p0 = wswz + (((size_t)kt * NT + nt) * 32 + n) * 16;
    H8 q0, q1, q2, q3;
    q0.f = *(const float4*)(p0);        // K = kt*32 + 0..7
    q1.f = *(const float4*)(p0 + 8);    // K = kt*32 + 8..15
    q2.f = *(const float4*)(p0 + 256);  // slot n+16: K = kt*32 + 16..23
    q3.f = *(const float4*)(p0 + 264);  // K = kt*32 + 24..31
    const float* av = in + kt * 32;
#pragma unroll
    for (int e = 0; e < 8; ++e) {
      acc += av[e]      * (float)q0.h[e];
      acc += av[8 + e]  * (float)q1.h[e];
      acc += av[16 + e] * (float)q2.h[e];
      acc += av[24 + e] * (float)q3.h[e];
    }
  }
  return acc;
}

struct Params {
  const float* intervals;
  const float* logsig;
  const float* b1;
  const float* b2;
  const float* b3;
  const _Float16* w1s;
  const _Float16* w2s;
  const _Float16* w3s;
  int T;
};

// ---------------------------------------------------------------------------
// One vector-field evaluation: kout = field(t, yv). Whole workgroup (512 thr).
// ---------------------------------------------------------------------------
__device__ void field_eval(float t, const float* yv, float* kout, const Params& P,
                           float* ls_s, float* Acoef,
                           float* a1, float* sd1, float* a2, float* sd2,
                           float* Wf, float* td3, float* res_s,
                           _Float16* Wh, _Float16* T1, _Float16* T2,
                           int* sh_idx, float* sh_den) {
  const int tid  = (int)threadIdx.x;
  const int lane = tid & 31;
  const int wave = tid >> 5;

  // --- P1: interval lookup (guess + narrow lower_bound over intervals[1:]) ---
  if (tid == 0) {
    int len = P.T - 1;
    float g = t * (float)len;
    int lo = (int)g - 8; if (lo < 0) lo = 0;
    int hi = lo + 16;    if (hi > len) hi = len;
    while (lo < hi) {
      int mid = (lo + hi) >> 1;
      if (P.intervals[mid + 1] < t) lo = mid + 1; else hi = mid;
    }
    int idx = lo + 1;
    if (idx < 1) idx = 1;
    if (idx > P.T - 1) idx = P.T - 1;
    *sh_idx = idx;
    *sh_den = P.intervals[idx] - P.intervals[idx - 1];
  }
  PHASE_SYNC();
  const int idx = *sh_idx;
  const float* ls = P.logsig + (size_t)(idx - 1) * LSIG_N;

  // --- P2: logsig row copy  ||  primal layer 1 (threads 256..511) ---
  if (tid < LSIG_N) ls_s[tid] = ls[tid];
  if (tid >= 256) {
    int c = tid - 256;
    float z = dotcol(P.w1s, yv, H_DIM, WID / 16, c) + P.b1[c];
    float sg = 1.f / (1.f + __expf(-z));
    a1[c]  = z * sg;
    sd1[c] = sg * (1.f + z * (1.f - sg));
  }
  PHASE_SYNC();

  // --- P3: antisymmetric bracket coefficients  ||  primal layer 2 ---
  if (tid < 256) {
    int i = tid >> 4, j = tid & 15;
    float v = 0.f;
    if (i < j)      { int p = i * 15 - (i * (i - 1)) / 2 + (j - i - 1); v =  ls_s[D_DIM + 1 + p]; }
    else if (i > j) { int p = j * 15 - (j * (j - 1)) / 2 + (i - j - 1); v = -ls_s[D_DIM + 1 + p]; }
    Acoef[tid] = v;
  } else {
    int c = tid - 256;
    float z = dotcol(P.w2s, a1, WID, WID / 16, c) + P.b2[c];
    float sg = 1.f / (1.f + __expf(-z));
    a2[c]  = z * sg;
    sd2[c] = sg * (1.f + z * (1.f - sg));
  }
  PHASE_SYNC();

  // --- P4: primal layer 3 -> W = tanh(z3), tanh' ; 4 cols/thread.
  //          Also zero the bracket accumulator for P7. ---
  if (tid < H_DIM) res_s[tid] = 0.f;
#pragma unroll 1
  for (int c = tid; c < OUTD; c += NTHREADS) {
    float z = dotcol(P.w3s, a2, WID, OUTD / 16, c) + P.b3[c];
    float w = tanhf(z);
    Wf[c]  = w;
    Wh[c]  = (_Float16)w;
    td3[c] = 1.f - w * w;
  }
  PHASE_SYNC();

  // --- P5: tangent layer 1 (WMMA, all 16 waves; 16 tiles) ---
  {
    int nt = wave;
    v8f acc = tile_mm(P.w1s, Wh, H_DIM, WID / 16, nt, lane);
    int n = lane & 15, col = nt * 16 + n, mb = (lane < 16) ? 0 : 8;
    float s = sd1[col];
#pragma unroll
    for (int r = 0; r < 8; ++r) T1[(mb + r) * WID + col] = (_Float16)(acc[r] * s);
  }
  PHASE_SYNC();

  // --- P6: tangent layer 2 (WMMA, all 16 waves; 16 tiles) ---
  {
    int nt = wave;
    v8f acc = tile_mm(P.w2s, T1, WID, WID / 16, nt, lane);
    int n = lane & 15, col = nt * 16 + n, mb = (lane < 16) ? 0 : 8;
    float s = sd2[col];
#pragma unroll
    for (int r = 0; r < 8; ++r) T2[(mb + r) * WID + col] = (_Float16)(acc[r] * s);
  }
  PHASE_SYNC();

  // --- P7: tangent layer 3 (WMMA, 8 tiles/wave) fused with Lie contraction ---
  // tile nt covers cols nt*16..+15 -> j = nt>>3 (all 16 share j), h = (nt&7)*16+n
#pragma unroll 1
  for (int nt = wave; nt < OUTD / 16; nt += 16) {
    v8f acc = tile_mm(P.w3s, T2, WID, OUTD / 16, nt, lane);
    int n = lane & 15, col = nt * 16 + n, mb = (lane < 16) ? 0 : 8;
    float td = td3[col];
    int j = nt >> 3, hb = (nt & 7) * 16;
    float s = 0.f;
#pragma unroll
    for (int r = 0; r < 8; ++r) s += Acoef[(mb + r) * 16 + j] * (acc[r] * td);
    atomicAdd(&res_s[hb + n], s);   // ds_add_f32
  }
  PHASE_SYNC();

  // --- P8: add linear term ls1 @ W and scale by interval length ---
  if (tid < H_DIM) {
    float s = res_s[tid];
#pragma unroll
    for (int k = 0; k < D_DIM; ++k) s += ls_s[1 + k] * Wf[k * H_DIM + tid];
    kout[tid] = s / *sh_den;
  }
  PHASE_SYNC();
}

// ---------------------------------------------------------------------------
// Persistent single-WGP kernel: 100 Heun steps + softmax head.
// ---------------------------------------------------------------------------
__global__ void __launch_bounds__(NTHREADS)
lncde_kernel(const float* __restrict__ ts, const float* __restrict__ logsig,
             const float* __restrict__ x0, const float* __restrict__ intervals,
             const float* __restrict__ l1w, const float* __restrict__ l1b,
             const float* __restrict__ b1, const float* __restrict__ b2,
             const float* __restrict__ b3, const float* __restrict__ l2w,
             const float* __restrict__ l2b,
             const _Float16* __restrict__ w1s, const _Float16* __restrict__ w2s,
             const _Float16* __restrict__ w3s,
             float* __restrict__ out, int T) {
  __shared__ float y_s[H_DIM], yt_s[H_DIM], k1_s[H_DIM], k2_s[H_DIM];
  __shared__ float ls_s[LSIG_N + 3];
  __shared__ float Acoef[256];
  __shared__ float a1[WID], sd1[WID], a2[WID], sd2[WID];
  __shared__ float Wf[OUTD], td3[OUTD], res_s[H_DIM];
  __shared__ _Float16 Wh[OUTD]      __attribute__((aligned(16)));
  __shared__ _Float16 T1[16 * WID]  __attribute__((aligned(16)));
  __shared__ _Float16 T2[16 * WID]  __attribute__((aligned(16)));
  __shared__ int   sh_idx;
  __shared__ float sh_den;

  const int tid = (int)threadIdx.x;
  Params P{intervals, logsig, b1, b2, b3, w1s, w2s, w3s, T};

  // y0 = x0 @ l1_w + l1_b
  if (tid < H_DIM) {
    float s = l1b[tid];
#pragma unroll
    for (int d = 0; d < D_DIM; ++d) s += x0[d] * l1w[d * H_DIM + tid];
    y_s[tid] = s;
  }
  float t  = ts[0];
  float dt = (ts[T - 1] - ts[0]) / (float)NSTEPS;
  PHASE_SYNC();

#pragma unroll 1
  for (int step = 0; step < NSTEPS; ++step) {
    field_eval(t, y_s, k1_s, P, ls_s, Acoef, a1, sd1, a2, sd2,
               Wf, td3, res_s, Wh, T1, T2, &sh_idx, &sh_den);
    if (tid < H_DIM) yt_s[tid] = y_s[tid] + dt * k1_s[tid];
    PHASE_SYNC();
    field_eval(t + dt, yt_s, k2_s, P, ls_s, Acoef, a1, sd1, a2, sd2,
               Wf, td3, res_s, Wh, T1, T2, &sh_idx, &sh_den);
    if (tid < H_DIM) y_s[tid] += 0.5f * dt * (k1_s[tid] + k2_s[tid]);
    PHASE_SYNC();
    t += dt;
  }

  // softmax(yT @ l2_w + l2_b)
  if (tid == 0) {
    float logits[10];
    float m = -3.4e38f;
    for (int j = 0; j < 10; ++j) {
      float s = l2b[j];
      for (int h = 0; h < H_DIM; ++h) s += y_s[h] * l2w[h * 10 + j];
      logits[j] = s;
      if (s > m) m = s;
    }
    float den = 0.f;
    for (int j = 0; j < 10; ++j) { logits[j] = __expf(logits[j] - m); den += logits[j]; }
    for (int j = 0; j < 10; ++j) out[j] = logits[j] / den;
  }
}

extern "C" void kernel_launch(void* const* d_in, const int* in_sizes, int n_in,
                              void* d_out, int out_size, void* d_ws, size_t ws_size,
                              hipStream_t stream) {
  (void)n_in; (void)out_size; (void)ws_size;
  const float* ts        = (const float*)d_in[0];
  const float* logsig    = (const float*)d_in[1];
  const float* x0        = (const float*)d_in[2];
  const float* intervals = (const float*)d_in[3];
  const float* l1w       = (const float*)d_in[4];
  const float* l1b       = (const float*)d_in[5];
  const float* w1        = (const float*)d_in[6];
  const float* b1        = (const float*)d_in[7];
  const float* w2        = (const float*)d_in[8];
  const float* b2        = (const float*)d_in[9];
  const float* w3        = (const float*)d_in[10];
  const float* b3        = (const float*)d_in[11];
  const float* l2w       = (const float*)d_in[12];
  const float* l2b       = (const float*)d_in[13];
  const int T = in_sizes[0];

  _Float16* w1s = (_Float16*)d_ws;          // 128*256 halves
  _Float16* w2s = w1s + 128 * 256;          // 256*256 halves
  _Float16* w3s = w2s + 256 * 256;          // 256*2048 halves (~1 MB)

  swz_kernel<<<256, 256, 0, stream>>>(w1, w1s, 128, 256);
  swz_kernel<<<256, 256, 0, stream>>>(w2, w2s, 256, 256);
  swz_kernel<<<512, 256, 0, stream>>>(w3, w3s, 256, 2048);
  lncde_kernel<<<1, NTHREADS, 0, stream>>>(ts, logsig, x0, intervals, l1w, l1b,
                                           b1, b2, b3, l2w, l2b,
                                           w1s, w2s, w3s, (float*)d_out, T);
}
